// MultiAttention_9010841387060
// MI455X (gfx1250) — matrix operand order repaired and checked by reading, not compiled, not run on hardware
//
#include <hip/hip_runtime.h>
#include <hip/hip_bf16.h>

typedef __bf16 bf16_t;
typedef bf16_t v16bf __attribute__((ext_vector_type(16)));
typedef bf16_t v8bf  __attribute__((ext_vector_type(8)));
typedef float  v8f   __attribute__((ext_vector_type(8)));

#define NH 16      // heads
#define HS 64      // head size
#define DM 1024    // model dim
#define NB 4       // batch
#define SL 2048    // sequence length

__device__ __forceinline__ v8f wmma_bf16(v16bf a, v16bf b, v8f c) {
  // v_wmma_f32_16x16x32_bf16: D = A(16x32) * B(32x16) + C(16x16 f32)
  return __builtin_amdgcn_wmma_f32_16x16x32_bf16(false, a, false, b, (short)0, c,
                                                 false, false);
}

__device__ __forceinline__ v16bf join8(v8bf lo, v8bf hi) {
  v16bf a;
#pragma unroll
  for (int i = 0; i < 8; ++i) { a[i] = lo[i]; a[i + 8] = hi[i]; }
  return a;
}

// gfx1250 async global->LDS copy (ASYNCcnt-tracked, bypasses VGPRs).
__device__ __forceinline__ void async_ld_b128(unsigned lds_addr, const void* gaddr) {
  asm volatile("global_load_async_to_lds_b128 %0, %1, off"
               :: "v"(lds_addr), "v"((unsigned long long)(uintptr_t)gaddr)
               : "memory");
}
__device__ __forceinline__ void wait_async4() {
  asm volatile("s_wait_asynccnt 0x4" ::: "memory");
}
__device__ __forceinline__ void wait_async0() {
  asm volatile("s_wait_asynccnt 0x0" ::: "memory");
}

// ---------------------------------------------------------------- weight cvt
__global__ void cvt_f32_to_bf16(const float* __restrict__ src,
                                bf16_t* __restrict__ dst, int n) {
  int i = blockIdx.x * 256 + threadIdx.x;
  if (i < n) dst[i] = (bf16_t)src[i];
}

// ---------------------------------------------------------------- projection
// One warp: one (matrix m, batch b, head, 16-token tile).
// out[t][o] = sum_h X[t][h] * Wv[o][h]   (Wv row-major [o][h] == column-major B)
// m=0 -> Qp [b][head][tok][64], m=1 -> Kp same, m=2 -> Vt [b][head][64][tok]
__global__ __launch_bounds__(128) void proj_kernel(
    const float* __restrict__ Qf, const float* __restrict__ Kf,
    const float* __restrict__ Vf, const bf16_t* __restrict__ Wv,
    bf16_t* __restrict__ Qp, bf16_t* __restrict__ Kp, bf16_t* __restrict__ Vt) {
  const int lane = threadIdx.x & 31;
  const int warp = threadIdx.x >> 5;
  const int ln = lane & 15;   // M (A) / N (B,C) index
  const int kh = lane >> 4;   // K-half selector

  int gw = blockIdx.x * 4 + warp;     // 24576 warps total
  const int head = gw & 15;  gw >>= 4;
  const int tt   = gw & 127; gw >>= 7;  // token tile (of 16)
  const int b    = gw & 3;   gw >>= 2;
  const int m    = gw;                  // 0=Q 1=K 2=V
  const float* X = (m == 0) ? Qf : (m == 1) ? Kf : Vf;

  const float* xrow = X + ((size_t)b * SL + tt * 16 + ln) * DM + head * HS;

  v8f acc[4];
#pragma unroll
  for (int j = 0; j < 4; ++j)
#pragma unroll
    for (int r = 0; r < 8; ++r) acc[j][r] = 0.0f;

#pragma unroll
  for (int s = 0; s < 2; ++s) {  // K-dim (input head dim 64) in 2 steps of 32
    const float* p0 = xrow + s * 32 + kh * 16;
    const float* p1 = p0 + 8;
    v16bf a;
#pragma unroll
    for (int i = 0; i < 8; ++i) { a[i] = (bf16_t)p0[i]; a[i + 8] = (bf16_t)p1[i]; }
#pragma unroll
    for (int j = 0; j < 4; ++j) {  // 4 N-tiles of output dim 64
      const bf16_t* wb = Wv + (size_t)(j * 16 + ln) * HS + s * 32 + kh * 16;
      v16bf bf = *(const v16bf*)wb;
      acc[j] = wmma_bf16(a, bf, acc[j]);
    }
  }

  if (m < 2) {
    bf16_t* dst = ((m == 0) ? Qp : Kp) + (((size_t)b * NH + head) * SL) * HS;
#pragma unroll
    for (int j = 0; j < 4; ++j)
#pragma unroll
      for (int r = 0; r < 8; ++r) {
        int tok = tt * 16 + r + 8 * kh;            // C layout: M = r + 8*half
        dst[(size_t)tok * HS + j * 16 + ln] = (bf16_t)acc[j][r];
      }
  } else {
    bf16_t* dst = Vt + (((size_t)b * NH + head) * HS) * SL;
#pragma unroll
    for (int j = 0; j < 4; ++j) {
      int h = j * 16 + ln;
      v8bf pk;
#pragma unroll
      for (int r = 0; r < 8; ++r) pk[r] = (bf16_t)acc[j][r];
      *(v8bf*)(dst + (size_t)h * SL + tt * 16 + 8 * kh) = pk;  // 16B store
    }
  }
}

// ---------------------------------------------------------------- attention
// Block = 4 warps sharing (b, head) with adjacent q-tiles. K'/V' chunks are
// cooperatively staged into LDS with async global->LDS copies (double
// buffered); each warp runs online softmax over 64 chunks of 32 keys.
// Reference quirks preserved: mask==0 -> -1e-20 before scale, scale = 1/32.
struct AttnSmem {
  bf16_t K[2][32][64];   // [buf][key tok][head dim]   8 KB
  bf16_t V[2][64][32];   // [buf][head dim][key tok]   8 KB
  bf16_t P[4][16][32];   // per-warp probability tile  4 KB
};

__global__ __launch_bounds__(128) void attn_kernel(
    const bf16_t* __restrict__ Qp, const bf16_t* __restrict__ Kp,
    const bf16_t* __restrict__ Vt, const int* __restrict__ mask,
    bf16_t* __restrict__ AO) {
  __shared__ __align__(16) AttnSmem sm;

  const int tid = threadIdx.x;
  const int lane = tid & 31;
  const int warp = tid >> 5;
  const int ln = lane & 15;
  const int kh = lane >> 4;

  int gw = blockIdx.x * 4 + warp;     // 8192 warps
  const int qt   = gw & 127; gw >>= 7;
  const int head = gw & 15;  gw >>= 4;
  const int b    = gw;

  const bf16_t* kbase = Kp + (((size_t)b * NH + head) * SL) * HS;
  const bf16_t* vbase = Vt + (((size_t)b * NH + head) * HS) * SL;
  const int* mbase = mask + (size_t)b * SL * SL;

  // ---- Q A-fragments (loop invariant)
  const bf16_t* qrow = Qp + (((size_t)b * NH + head) * SL + qt * 16 + ln) * HS;
  v16bf aq[2];
#pragma unroll
  for (int s = 0; s < 2; ++s) {
    v8bf lo = *(const v8bf*)(qrow + s * 32 + kh * 16);
    v8bf hi = *(const v8bf*)(qrow + s * 32 + kh * 16 + 8);
    aq[s] = join8(lo, hi);
  }

  // ---- cooperative async stage of one 32-key chunk into LDS buffer `buf`
  // K chunk: 32*64 bf16 contiguous (4 KB); V chunk: 64 rows of 32 tok (64 B).
  auto stage = [&](int buf, int kb) {
    const char* kg = (const char*)(kbase + (size_t)kb * HS) + tid * 32;
    unsigned kl = (unsigned)(uintptr_t)&sm.K[buf][0][0] + tid * 32;
    async_ld_b128(kl, kg);
    async_ld_b128(kl + 16, kg + 16);
    int h = tid >> 1, half = tid & 1;
    const char* vg = (const char*)(vbase + (size_t)h * SL + kb) + half * 32;
    unsigned vl = (unsigned)(uintptr_t)&sm.V[buf][h][0] + half * 32;
    async_ld_b128(vl, vg);
    async_ld_b128(vl + 16, vg + 16);
  };

  v8f O[4];
#pragma unroll
  for (int j = 0; j < 4; ++j)
#pragma unroll
    for (int r = 0; r < 8; ++r) O[j][r] = 0.0f;
  float mrun[8], lrun[8];
#pragma unroll
  for (int r = 0; r < 8; ++r) { mrun[r] = -__builtin_inff(); lrun[r] = 0.0f; }

  const float scl = 0.03125f;  // 1/sqrt(1024)

  stage(0, 0);  // prefetch first chunk

  for (int kb = 0; kb < SL; kb += 32) {
    const int cur = (kb >> 5) & 1;
    const int nxt = cur ^ 1;
    const bool has_next = (kb + 32) < SL;
    if (has_next) {
      stage(nxt, kb + 32);  // overlap next chunk's DMA with current compute
      wait_async4();        // in-order completion => current chunk resident
    } else {
      wait_async0();
    }
    __syncthreads();  // publish staged K/V; also WAR vs prev iter's P loads

    // ---- S = Q' K'^T (two 16-wide N tiles) from LDS
    v8f S0, S1;
#pragma unroll
    for (int r = 0; r < 8; ++r) { S0[r] = 0.0f; S1[r] = 0.0f; }
#pragma unroll
    for (int s = 0; s < 2; ++s) {
      v16bf bk0 = *(const v16bf*)(&sm.K[cur][ln][0] + s * 32 + kh * 16);
      v16bf bk1 = *(const v16bf*)(&sm.K[cur][16 + ln][0] + s * 32 + kh * 16);
      S0 = wmma_bf16(aq[s], bk0, S0);
      S1 = wmma_bf16(aq[s], bk1, S1);
    }

    // ---- mask, scale, online softmax (row r+8*kh lives in 16-lane half)
    float p0v[8], p1v[8];
#pragma unroll
    for (int r = 0; r < 8; ++r) {
      int q = qt * 16 + r + 8 * kh;
      const int* mr = mbase + (size_t)q * SL + kb;
      int mm0 = mr[ln], mm1 = mr[16 + ln];
      float z0 = (mm0 == 0) ? -1e-20f : S0[r];
      float z1 = (mm1 == 0) ? -1e-20f : S1[r];
      z0 *= scl; z1 *= scl;
      float mx = fmaxf(z0, z1);
#pragma unroll
      for (int off = 8; off >= 1; off >>= 1) mx = fmaxf(mx, __shfl_xor(mx, off));
      float mn = fmaxf(mrun[r], mx);
      float p0 = __expf(z0 - mn);
      float p1 = __expf(z1 - mn);
      float ps = p0 + p1;
#pragma unroll
      for (int off = 8; off >= 1; off >>= 1) ps += __shfl_xor(ps, off);
      float fac = __expf(mrun[r] - mn);
      lrun[r] = lrun[r] * fac + ps;
      mrun[r] = mn;
#pragma unroll
      for (int j = 0; j < 4; ++j) O[j][r] *= fac;
      p0v[r] = p0; p1v[r] = p1;
    }

    // ---- C-layout -> A-layout reshape of P through LDS
#pragma unroll
    for (int r = 0; r < 8; ++r) {
      sm.P[warp][r + 8 * kh][ln]      = (bf16_t)p0v[r];
      sm.P[warp][r + 8 * kh][16 + ln] = (bf16_t)p1v[r];
    }
    __syncthreads();
    const bf16_t* pr = &sm.P[warp][ln][0];
    v8bf plo = *(const v8bf*)(pr + kh * 16);
    v8bf phi = *(const v8bf*)(pr + 8 + kh * 16);
    v16bf pa = join8(plo, phi);

    // ---- O += P * V' from LDS (B operand K-contiguous)
#pragma unroll
    for (int j = 0; j < 4; ++j) {
      v16bf bv = *(const v16bf*)(&sm.V[cur][j * 16 + ln][0] + kh * 16);
      O[j] = wmma_bf16(pa, bv, O[j]);
    }
    __syncthreads();  // reads of buffer `cur` done before it is re-staged
  }

  // ---- finalize: divide by softmax denom, store [b][q][head*64+h] bf16
#pragma unroll
  for (int r = 0; r < 8; ++r) {
    float inv = 1.0f / lrun[r];
    int q = qt * 16 + r + 8 * kh;
    bf16_t* orow = AO + ((size_t)b * SL + q) * DM + head * HS;
#pragma unroll
    for (int j = 0; j < 4; ++j) orow[j * 16 + ln] = (bf16_t)(O[j][r] * inv);
  }
}

// ---------------------------------------------------------------- W_o proj
// out[t][o] = sum_d AO[t][d] * Wo[o][d]; one warp: 16 tokens x 64 outputs.
__global__ __launch_bounds__(128) void oproj_kernel(
    const bf16_t* __restrict__ AO, const bf16_t* __restrict__ Wo,
    float* __restrict__ out) {
  const int lane = threadIdx.x & 31;
  const int warp = threadIdx.x >> 5;
  const int ln = lane & 15;
  const int kh = lane >> 4;

  int gw = blockIdx.x * 4 + warp;     // 8192 warps
  const int ot = gw & 15; gw >>= 4;   // 64-wide output tile
  const int tt = gw;                  // 16-token tile (0..511)

  const bf16_t* arow = AO + (size_t)(tt * 16 + ln) * DM;

  v8f acc[4];
#pragma unroll
  for (int j = 0; j < 4; ++j)
#pragma unroll
    for (int r = 0; r < 8; ++r) acc[j][r] = 0.0f;

  for (int kb = 0; kb < DM; kb += 32) {
    v8bf lo = *(const v8bf*)(arow + kb + kh * 16);
    v8bf hi = *(const v8bf*)(arow + kb + kh * 16 + 8);
    v16bf a = join8(lo, hi);
#pragma unroll
    for (int j = 0; j < 4; ++j) {
      const bf16_t* wb = Wo + (size_t)(ot * 64 + j * 16 + ln) * DM + kb + kh * 16;
      v16bf bf = *(const v16bf*)wb;
      acc[j] = wmma_bf16(a, bf, acc[j]);
    }
  }

#pragma unroll
  for (int j = 0; j < 4; ++j)
#pragma unroll
    for (int r = 0; r < 8; ++r) {
      int tok = tt * 16 + r + 8 * kh;
      out[(size_t)tok * DM + ot * 64 + j * 16 + ln] = acc[j][r];
    }
}

// ---------------------------------------------------------------- launcher
extern "C" void kernel_launch(void* const* d_in, const int* in_sizes, int n_in,
                              void* d_out, int out_size, void* d_ws, size_t ws_size,
                              hipStream_t stream) {
  const float* keys    = (const float*)d_in[0];
  const float* values  = (const float*)d_in[1];
  const float* queries = (const float*)d_in[2];
  const int*   mask    = (const int*)d_in[3];
  const float* W_v     = (const float*)d_in[4];
  const float* W_o     = (const float*)d_in[5];
  float* out = (float*)d_out;

  // workspace layout (bytes)
  const size_t szWv  = (size_t)HS * HS * 2;           // 8 KB
  const size_t szWo  = (size_t)DM * DM * 2;           // 2 MB
  const size_t szQKV = (size_t)NB * NH * SL * HS * 2; // 16.78 MB each
  char* ws = (char*)d_ws;
  bf16_t* wsWv = (bf16_t*)(ws);
  bf16_t* wsWo = (bf16_t*)(ws + szWv);
  bf16_t* Qp   = (bf16_t*)(ws + szWv + szWo);
  bf16_t* Kp   = (bf16_t*)(ws + szWv + szWo + szQKV);
  bf16_t* Vt   = (bf16_t*)(ws + szWv + szWo + 2 * szQKV);
  bf16_t* AO   = (bf16_t*)(ws + szWv + szWo + 3 * szQKV);

  cvt_f32_to_bf16<<<(HS * HS + 255) / 256, 256, 0, stream>>>(W_v, wsWv, HS * HS);
  cvt_f32_to_bf16<<<(DM * DM + 255) / 256, 256, 0, stream>>>(W_o, wsWo, DM * DM);

  // 3 matrices * 4 batches * 128 token-tiles * 16 heads = 24576 warps / 4 blocks
  proj_kernel<<<6144, 128, 0, stream>>>(queries, keys, values, wsWv, Qp, Kp, Vt);

  // 4 * 16 * 128 = 8192 warps / 4 = 2048 blocks
  attn_kernel<<<2048, 128, 0, stream>>>(Qp, Kp, Vt, mask, AO);

  // 512 token-tiles * 16 o-tiles = 8192 warps / 4 = 2048 blocks
  oproj_kernel<<<2048, 128, 0, stream>>>(AO, wsWo, out);
}
